// NonLocal1D_32212254720349
// MI455X (gfx1250) — matrix operand order repaired and checked by reading, not compiled, not run on hardware
//
#include <hip/hip_runtime.h>
#include <hip/hip_bf16.h>

// ---------------------------------------------------------------------------
// NonLocal1D block for MI455X (gfx1250, wave32, WMMA f16 16x16x32).
// Pipeline: prep -> transpose(x) -> proj(theta/phi/g) -> flash-attn -> zgemm
//           -> bn finalize -> elementwise output.
// ---------------------------------------------------------------------------

typedef _Float16 v8h  __attribute__((ext_vector_type(8)));
typedef _Float16 v16h __attribute__((ext_vector_type(16)));
typedef float    v8f  __attribute__((ext_vector_type(8)));

static constexpr int B_  = 8;
static constexpr int C_  = 512;
static constexpr int CI  = 256;   // C/2
static constexpr int L_  = 4096;
static constexpr int LK  = 2048;  // L/2 after maxpool2
static constexpr int NW  = CI * C_;  // 131072 elems per projection weight

// Workspace layout (bytes). Region [0,64MB) first holds xT/thetaT/phiT/gN
// (all dead after attention), then is reused for z (f32, 64MB).
static constexpr size_t XT_OFF     = 0;                          // [L,C] f16 per batch   32MB
static constexpr size_t THETAT_OFF = 33554432;                   // [L,Ci] f16            16MB
static constexpr size_t PHIT_OFF   = 50331648;                   // [Lk,Ci] f16            8MB
static constexpr size_t GN_OFF     = 58720256;                   // [Ci,Lk] f16            8MB
static constexpr size_t Z_OFF      = 0;                          // [C,L] f32 (overlay)   64MB
static constexpr size_t YT_OFF     = 67108864;                   // [L,Ci] f16            16MB
static constexpr size_t WH_OFF     = 83886080;                   // 4*NW f16               1MB
static constexpr size_t STATS_OFF  = 84934656;                   // 4*C f32 (sum,sq,scale,shift)

// --------------------------- WMMA helpers ----------------------------------

__device__ inline v8f wmma_f16(v16h a, v16h b, v8f c) {
  // D = A(16x32 f16) * B(32x16 f16) + C(16x16 f32)
  return __builtin_amdgcn_wmma_f32_16x16x32_f16(false, a, false, b, (short)0, c,
                                                false, false);
}

// A-operand (16x32 f16): lane row holds K 0..15 (lanes 0-15) or
// 16..31 (lanes 16-31). base = &row[kb]; h = lane>>4.
__device__ inline v16h load_a16(const _Float16* __restrict__ base, int h) {
  return *(const v16h*)(base + 16 * h);
}

// B-operand (32x16 f16): lane column holds K 0..15 (lanes 0-15) or 16..31.
// base = &col[kb] with K contiguous in memory; h = lane>>4.
__device__ inline v16h load_b16(const _Float16* __restrict__ base, int h) {
  return *(const v16h*)(base + 16 * h);
}

// --------------------------- kernel 0: prep --------------------------------

__global__ void prep_kernel(const float* __restrict__ tw, const float* __restrict__ pw,
                            const float* __restrict__ gw, const float* __restrict__ zw,
                            _Float16* __restrict__ wh, float* __restrict__ stats) {
  int i = blockIdx.x * blockDim.x + threadIdx.x;
  if (i < NW)               wh[i] = (_Float16)tw[i];
  else if (i < 2 * NW)      wh[i] = (_Float16)pw[i - NW];
  else if (i < 3 * NW)      wh[i] = (_Float16)gw[i - 2 * NW];
  else if (i < 4 * NW)      wh[i] = (_Float16)zw[i - 3 * NW];
  if (i < 4 * C_) stats[i] = 0.0f;
}

// ----------------------- kernel 1: transpose x -> f16 ----------------------

__global__ void transpose_x_kernel(const float* __restrict__ x, _Float16* __restrict__ xT) {
  __shared__ _Float16 tile[32][33];
  int b  = blockIdx.z;
  int c0 = blockIdx.y * 32;
  int l0 = blockIdx.x * 32;
  const float* xb = x + (size_t)b * C_ * L_;
  _Float16* xTb = xT + (size_t)b * L_ * C_;
  int tx = threadIdx.x, ty = threadIdx.y;
#pragma unroll
  for (int i = 0; i < 4; ++i) {
    int c = c0 + ty + i * 8;
    tile[ty + i * 8][tx] = (_Float16)xb[(size_t)c * L_ + l0 + tx];
  }
  __syncthreads();
#pragma unroll
  for (int i = 0; i < 4; ++i) {
    int l = l0 + ty + i * 8;
    xTb[(size_t)l * C_ + c0 + tx] = tile[tx][ty + i * 8];
  }
}

// -------------------- kernel 2: projections (WMMA GEMM) --------------------
// out[co,l] = W[co,:] . x[:,l] + bias[co]
// MODE 0: theta -> thetaT [L,Ci] (row q, contiguous c)
// MODE 1: phi   -> maxpool2 over l -> phiT [Lk,Ci] (row k, contiguous c)
// MODE 2: g     -> maxpool2 over l -> gN [Ci,Lk]  (row c, contiguous k)

template <int MODE>
__global__ __launch_bounds__(32) void proj_kernel(const _Float16* __restrict__ xT,
                                                  const _Float16* __restrict__ W,
                                                  const float* __restrict__ bias,
                                                  _Float16* __restrict__ out) {
  const int lane = threadIdx.x;
  const int h = lane >> 4, ln = lane & 15;
  const int b = blockIdx.z;
  const int co0 = blockIdx.y * 16;
  const int NT = (MODE == 0) ? 16 : 32;
  const int l0 = blockIdx.x * NT;
  const _Float16* xTb = xT + (size_t)b * L_ * C_;
  const _Float16* wrow = W + (size_t)(co0 + ln) * C_;

  v8f acc0 = {}, acc1 = {};
#pragma unroll
  for (int cb = 0; cb < C_; cb += 32) {
    v16h a  = load_a16(wrow + cb, h);
    v16h b0 = load_b16(xTb + (size_t)(l0 + ln) * C_ + cb, h);
    acc0 = wmma_f16(a, b0, acc0);
    if (MODE != 0) {
      v16h b1 = load_b16(xTb + (size_t)(l0 + 16 + ln) * C_ + cb, h);
      acc1 = wmma_f16(a, b1, acc1);
    }
  }
  // bias: D-layout row co = co0 + r + 8h
#pragma unroll
  for (int r = 0; r < 8; ++r) {
    float bb = bias[co0 + r + 8 * h];
    acc0[r] += bb;
    acc1[r] += bb;
  }

  if (MODE == 0) {
    _Float16* ob = out + (size_t)b * L_ * CI;
    v8h v;
#pragma unroll
    for (int r = 0; r < 8; ++r) v[r] = (_Float16)acc0[r];
    *(v8h*)(ob + (size_t)(l0 + ln) * CI + co0 + 8 * h) = v;
  } else {
    // maxpool2 over adjacent columns (adjacent lanes within 16-group)
    float p0[8], p1[8];
#pragma unroll
    for (int r = 0; r < 8; ++r) {
      p0[r] = fmaxf(acc0[r], __shfl_xor(acc0[r], 1, 32));
      p1[r] = fmaxf(acc1[r], __shfl_xor(acc1[r], 1, 32));
    }
    if ((lane & 1) == 0) {
      int j = ln >> 1;
      int kbase = l0 >> 1;
      if (MODE == 1) {
        _Float16* ob = out + (size_t)b * LK * CI;
        v8h v0, v1;
#pragma unroll
        for (int r = 0; r < 8; ++r) { v0[r] = (_Float16)p0[r]; v1[r] = (_Float16)p1[r]; }
        *(v8h*)(ob + (size_t)(kbase + j) * CI + co0 + 8 * h) = v0;
        *(v8h*)(ob + (size_t)(kbase + 8 + j) * CI + co0 + 8 * h) = v1;
      } else {
        _Float16* ob = out + (size_t)b * CI * LK;
#pragma unroll
        for (int r = 0; r < 8; ++r) {
          int co = co0 + r + 8 * h;
          ob[(size_t)co * LK + kbase + j]     = (_Float16)p0[r];
          ob[(size_t)co * LK + kbase + 8 + j] = (_Float16)p1[r];
        }
      }
    }
  }
}

// ---------------- kernel 3: flash attention (per 16-query tile) ------------
// S^T[k,q] = sum_c phi[c,k] theta[c,q]; online softmax over k; y = P.g
// P^T D-layout is re-used as the PV B-operand with one shfl_xor(16).

__global__ __launch_bounds__(32) void attn_kernel(const _Float16* __restrict__ thetaT,
                                                  const _Float16* __restrict__ phiT,
                                                  const _Float16* __restrict__ gN,
                                                  _Float16* __restrict__ yT) {
  const int lane = threadIdx.x;
  const int h = lane >> 4, ln = lane & 15;
  const int b = blockIdx.y;
  const int q0 = blockIdx.x * 16;

  const _Float16* thb = thetaT + (size_t)b * L_ * CI + (size_t)(q0 + ln) * CI;
  const _Float16* phb = phiT + (size_t)b * LK * CI;
  const _Float16* gb  = gN + (size_t)b * CI * LK;

  v8f zero8 = {};
  v8f acc[16];
#pragma unroll
  for (int ct = 0; ct < 16; ++ct) acc[ct] = zero8;

  float m_run = -1e30f, lsum = 0.0f;

  for (int kb = 0; kb < LK; kb += 32) {
    // --- S^T tiles: s0 covers k = kb..kb+15, s1 covers kb+16..kb+31
    v8f s0 = zero8, s1 = zero8;
#pragma unroll
    for (int cb = 0; cb < CI; cb += 32) {
      v16h bt = load_b16(thb + cb, h);
      v16h a0 = load_a16(phb + (size_t)(kb + ln) * CI + cb, h);
      s0 = wmma_f16(a0, bt, s0);
      v16h a1 = load_a16(phb + (size_t)(kb + 16 + ln) * CI + cb, h);
      s1 = wmma_f16(a1, bt, s1);
    }
    // --- online softmax over k (16 in-lane values + partner half)
    float mt = s0[0];
#pragma unroll
    for (int r = 1; r < 8; ++r) mt = fmaxf(mt, s0[r]);
#pragma unroll
    for (int r = 0; r < 8; ++r) mt = fmaxf(mt, s1[r]);
    mt = fmaxf(mt, __shfl_xor(mt, 16, 32));
    float m_new = fmaxf(m_run, mt);
    float corr = __expf(m_run - m_new);

    float p0[8], p1[8], rs = 0.0f;
#pragma unroll
    for (int r = 0; r < 8; ++r) {
      p0[r] = __expf(s0[r] - m_new);
      p1[r] = __expf(s1[r] - m_new);
      rs += p0[r] + p1[r];
    }
    rs += __shfl_xor(rs, 16, 32);
    lsum = lsum * corr + rs;
    m_run = m_new;

#pragma unroll
    for (int ct = 0; ct < 16; ++ct)
#pragma unroll
      for (int r = 0; r < 8; ++r) acc[ct][r] *= corr;

    // --- build PV B-operand (32x16 f16) from P^T D-layout
    // lanes 0-15 need k kb..kb+15 of column q; lanes 16-31 need kb+16..kb+31.
    v16h Bp;
#pragma unroll
    for (int r = 0; r < 8; ++r) {
      float o0 = __shfl_xor(p0[r], 16, 32);
      float o1 = __shfl_xor(p1[r], 16, 32);
      float lo = h ? o1 : p0[r];
      float hi = h ? p1[r] : o0;
      Bp[r] = (_Float16)lo;
      Bp[8 + r] = (_Float16)hi;
    }
    // --- PV: y[c,q] += g[c,kb..kb+31] * P^T
#pragma unroll
    for (int ct = 0; ct < 16; ++ct) {
      v16h Ag = load_a16(gb + (size_t)(ct * 16 + ln) * LK + kb, h);
      acc[ct] = wmma_f16(Ag, Bp, acc[ct]);
    }
  }

  float inv = 1.0f / lsum;
  _Float16* yrow = yT + (size_t)b * L_ * CI + (size_t)(q0 + ln) * CI + 8 * h;
#pragma unroll
  for (int ct = 0; ct < 16; ++ct) {
    v8h v;
#pragma unroll
    for (int r = 0; r < 8; ++r) v[r] = (_Float16)(acc[ct][r] * inv);
    *(v8h*)(yrow + ct * 16) = v;
  }
}

// ------------------- kernel 4: z = wz.y + b, BN stat accum -----------------

__global__ __launch_bounds__(32) void zgemm_kernel(const _Float16* __restrict__ yT,
                                                   const _Float16* __restrict__ Wz,
                                                   const float* __restrict__ zb,
                                                   float* __restrict__ z,
                                                   float* __restrict__ stats) {
  const int lane = threadIdx.x;
  const int h = lane >> 4, ln = lane & 15;
  const int b = blockIdx.z;
  const int co0 = blockIdx.y * 16;
  const int l0 = blockIdx.x * 16;
  const _Float16* yb = yT + (size_t)b * L_ * CI + (size_t)(l0 + ln) * CI;
  const _Float16* wrow = Wz + (size_t)(co0 + ln) * CI;

  v8f acc = {};
#pragma unroll
  for (int cb = 0; cb < CI; cb += 32) {
    v16h a  = load_a16(wrow + cb, h);
    v16h bv = load_b16(yb + cb, h);
    acc = wmma_f16(a, bv, acc);
  }

  float* zbp = z + (size_t)b * C_ * L_;
  float* sum = stats;
  float* sq  = stats + C_;
#pragma unroll
  for (int r = 0; r < 8; ++r) {
    int co = co0 + r + 8 * h;
    float v = acc[r] + zb[co];
    zbp[(size_t)co * L_ + l0 + ln] = v;
    float s = v, s2 = v * v;
    s  += __shfl_xor(s, 1, 32);  s2 += __shfl_xor(s2, 1, 32);
    s  += __shfl_xor(s, 2, 32);  s2 += __shfl_xor(s2, 2, 32);
    s  += __shfl_xor(s, 4, 32);  s2 += __shfl_xor(s2, 4, 32);
    s  += __shfl_xor(s, 8, 32);  s2 += __shfl_xor(s2, 8, 32);
    if (ln == 0) {
      atomicAdd(&sum[co], s);
      atomicAdd(&sq[co], s2);
    }
  }
}

// -------------------- kernel 5: BN fold to scale/shift ---------------------

__global__ void bnfin_kernel(float* __restrict__ stats, const float* __restrict__ gamma,
                             const float* __restrict__ beta) {
  int c = blockIdx.x * blockDim.x + threadIdx.x;
  if (c < C_) {
    float n = (float)B_ * (float)L_;
    float mean = stats[c] / n;
    float var = stats[C_ + c] / n - mean * mean;
    float rstd = rsqrtf(var + 1e-5f);
    float sc = gamma[c] * rstd;
    stats[2 * C_ + c] = sc;
    stats[3 * C_ + c] = beta[c] - mean * sc;
  }
}

// ------------------- kernel 6: out = z*scale + shift + x -------------------

__global__ void out_kernel(const float* __restrict__ z, const float* __restrict__ x,
                           const float* __restrict__ stats, float* __restrict__ out) {
  size_t i = (size_t)blockIdx.x * blockDim.x + threadIdx.x;
  int c = (int)((i / L_) % C_);
  out[i] = z[i] * stats[2 * C_ + c] + stats[3 * C_ + c] + x[i];
}

// ---------------------------------------------------------------------------

extern "C" void kernel_launch(void* const* d_in, const int* in_sizes, int n_in,
                              void* d_out, int out_size, void* d_ws, size_t ws_size,
                              hipStream_t stream) {
  const float* x        = (const float*)d_in[0];
  const float* theta_w  = (const float*)d_in[1];
  const float* theta_b  = (const float*)d_in[2];
  const float* phi_w    = (const float*)d_in[3];
  const float* phi_b    = (const float*)d_in[4];
  const float* g_w      = (const float*)d_in[5];
  const float* g_b      = (const float*)d_in[6];
  const float* wz_w     = (const float*)d_in[7];
  const float* wz_b     = (const float*)d_in[8];
  const float* bn_gamma = (const float*)d_in[9];
  const float* bn_beta  = (const float*)d_in[10];

  char* ws = (char*)d_ws;
  _Float16* xT     = (_Float16*)(ws + XT_OFF);
  _Float16* thetaT = (_Float16*)(ws + THETAT_OFF);
  _Float16* phiT   = (_Float16*)(ws + PHIT_OFF);
  _Float16* gN     = (_Float16*)(ws + GN_OFF);
  _Float16* yT     = (_Float16*)(ws + YT_OFF);
  _Float16* wh     = (_Float16*)(ws + WH_OFF);
  float*    z      = (float*)(ws + Z_OFF);   // overlays xT.. region (dead by then)
  float*    stats  = (float*)(ws + STATS_OFF);
  float*    out    = (float*)d_out;

  // 0) weight conversion + stat zeroing (re-done every launch: replay-safe)
  prep_kernel<<<dim3((4 * NW + 255) / 256), dim3(256), 0, stream>>>(
      theta_w, phi_w, g_w, wz_w, wh, stats);

  // 1) x [B,C,L] f32 -> xT [B,L,C] f16
  transpose_x_kernel<<<dim3(L_ / 32, C_ / 32, B_), dim3(32, 8), 0, stream>>>(x, xT);

  // 2) projections
  proj_kernel<0><<<dim3(L_ / 16, CI / 16, B_), dim3(32), 0, stream>>>(xT, wh, theta_b, thetaT);
  proj_kernel<1><<<dim3(L_ / 32, CI / 16, B_), dim3(32), 0, stream>>>(xT, wh + NW, phi_b, phiT);
  proj_kernel<2><<<dim3(L_ / 32, CI / 16, B_), dim3(32), 0, stream>>>(xT, wh + 2 * NW, g_b, gN);

  // 3) flash attention -> yT [B,L,Ci] f16
  attn_kernel<<<dim3(L_ / 16, B_), dim3(32), 0, stream>>>(thetaT, phiT, gN, yT);

  // 4) z = wz.y + b (f32) + per-channel sum/sumsq
  zgemm_kernel<<<dim3(L_ / 16, C_ / 16, B_), dim3(32), 0, stream>>>(yT, wh + 3 * NW, wz_b, z, stats);

  // 5) BN fold
  bnfin_kernel<<<dim3(2), dim3(256), 0, stream>>>(stats, bn_gamma, bn_beta);

  // 6) out = z_hat*gamma + beta + x
  out_kernel<<<dim3((B_ * C_ * L_) / 256), dim3(256), 0, stream>>>(z, x, stats, out);
}
